// SimpleRNN_55319178772869
// MI455X (gfx1250) — matrix operand order repaired and checked by reading, not compiled, not run on hardware
//
#include <hip/hip_runtime.h>
#include <hip/hip_bf16.h>
#include <math.h>

// Problem constants (from reference): B=1024, T=512, D=64, H=64, O=32, 4H=256
#define BB   1024
#define TT   512
#define DD   64
#define HH   64
#define OO   32
#define G4H  256

typedef __attribute__((ext_vector_type(16))) _Float16 v16h;
typedef __attribute__((ext_vector_type(8)))  _Float16 v8h;
typedef __attribute__((ext_vector_type(8)))  float    v8f;

union V16 {
    v16h v;
    v8h  h2[2];
    _Float16 e[16];
};

__device__ __forceinline__ float fsigmoid(float x) {
    return 1.0f / (1.0f + __expf(-x));
}
__device__ __forceinline__ float ftanh(float x) {
    float e = __expf(2.0f * x);
    return (e - 1.0f) / (e + 1.0f);
}

// ---------------------------------------------------------------------------
// Kernel 1: f32 -> f16 conversion + (B,T,D) -> (T,B,D) transpose of layer-0
// input. Both read and write are coalesced in 64-element runs.
// ---------------------------------------------------------------------------
__global__ void k_cvt_transpose(const float* __restrict__ src,
                                _Float16* __restrict__ dst) {
    size_t i = (size_t)blockIdx.x * 256 + threadIdx.x;  // output index
    int    d  = (int)(i & (DD - 1));
    size_t rt = i >> 6;                 // t*B + b
    int    b  = (int)(rt & (BB - 1));
    int    t  = (int)(rt >> 10);
    dst[i] = (_Float16)src[((size_t)b * TT + t) * DD + d];
}

// ---------------------------------------------------------------------------
// Kernel 2: pre = xin @ W_ih^T + (b_ih + b_hh), f32 accumulators dumped in
// raw WMMA C-fragment lane order so the recurrent kernel reloads them as
// v8f registers with zero rearrangement.
//   xin: (T*B, 64) f16, time-major (row = t*B + b)
//   pre: [t][btile(64)][ntile(16)][lane(32)][r(8)] f32
// Block: 256 threads (8 waves); wave owns one 16-row M-tile (fixed t,
// 16 consecutive b), loops over 16 N-tiles; K=64 -> 2 WMMA per tile.
// ---------------------------------------------------------------------------
__global__ void __launch_bounds__(256)
k_gemm_pre(const _Float16* __restrict__ xin,
           const float* __restrict__ w,
           const float* __restrict__ b_ih,
           const float* __restrict__ b_hh,
           float* __restrict__ pre) {
    __shared__ _Float16 wls[G4H][72];   // padded rows: stride 36 dwords
    __shared__ float    bls[G4H];

    const int tid  = threadIdx.x;
    const int wv   = tid >> 5;
    const int lane = tid & 31;
    const int g    = lane >> 4;
    const int n    = lane & 15;         // doubles as M-row index for A loads

    // Cooperative load of W (f32 -> f16) + combined bias into LDS.
    {
        const float* wr = w + tid * DD;
        #pragma unroll
        for (int k = 0; k < DD; ++k) wls[tid][k] = (_Float16)wr[k];
        bls[tid] = b_ih[tid] + b_hh[tid];
    }
    __syncthreads();

    const size_t rt0 = (size_t)blockIdx.x * 128 + wv * 16;  // tile base row
    const size_t row = rt0 + n;                             // A row (m = n)

    // A fragments: 16-bit A 16x32 layout. lane (g, m): halves 0..7 hold
    // K = kf*32 + g*8 + {0..7}; halves 8..15 hold K = kf*32 + 16 + g*8 + {0..7}
    V16 a[2];
    #pragma unroll
    for (int kf = 0; kf < 2; ++kf) {
        const _Float16* ar = xin + row * DD + kf * 32 + g * 8;
        a[kf].h2[0] = *(const v8h*)(ar);
        a[kf].h2[1] = *(const v8h*)(ar + 16);
    }

    // Output base: (t*64 + btile)*16 n-tiles
    const size_t t_   = rt0 >> 10;            // / B
    const size_t bt   = (rt0 & (BB - 1)) >> 4;
    const size_t base = (t_ * 64 + bt) * 16;

    #pragma unroll 4
    for (int nt = 0; nt < 16; ++nt) {
        // B fragments: 16-bit B 32x16 layout. lane (g, n): halves 0..7 hold
        // K = kf*32 + g*8 + {0..7}; halves 8..15 hold K = kf*32 + 16 + g*8 + {0..7}, value = W[col][K]; col = nt*16 + n.
        V16 bf[2];
        #pragma unroll
        for (int kf = 0; kf < 2; ++kf) {
            const _Float16* br = &wls[nt * 16 + n][kf * 32 + g * 8];
            bf[kf].h2[0] = *(const v8h*)(br);
            bf[kf].h2[1] = *(const v8h*)(br + 16);
        }

        v8f acc;
        const float bias = bls[nt * 16 + n];   // broadcast over M rows
        #pragma unroll
        for (int r = 0; r < 8; ++r) acc[r] = bias;

        acc = __builtin_amdgcn_wmma_f32_16x16x32_f16(
                false, a[0].v, false, bf[0].v, (short)0, acc, false, false);
        acc = __builtin_amdgcn_wmma_f32_16x16x32_f16(
                false, a[1].v, false, bf[1].v, (short)0, acc, false, false);

        // Raw fragment dump: one contiguous 32B store per lane. Streamed
        // non-temporally (pre = 512MB > 192MB L2).
        v8f* dst = (v8f*)(pre + (base + nt) * 256 + lane * 8);
        __builtin_nontemporal_store(acc, dst);
    }
}

// ---------------------------------------------------------------------------
// Kernel 3: LSTM recurrence for one layer.
//   pre : fragment-layout f32 (see k_gemm_pre)
//   w_hh: (256, 64) f32
//   hout: (T*B, 64) f16 time-major (row = t*B + b) -> next layer's GEMM input
// Grid: B/16 = 64 blocks, 128 threads (4 waves). Wave w owns hidden columns
// [16w, 16w+16); its gate columns are q*64 + 16w. W_hh B-fragments pinned in
// VGPRs for all 512 steps. pre tiles prefetched at distance 2 into ping-pong
// register sets. Per step: 8 x v_wmma_f32_16x16x32_f16 + gate VALU.
// ---------------------------------------------------------------------------
__device__ __forceinline__ v8f ld_pre(const float* __restrict__ pre,
                                      int bt, int wv, int lane, int t, int q) {
    return *(const v8f*)(pre + (((size_t)t * 64 + bt) * 16 + (q * 4 + wv)) * 256
                             + lane * 8);
}

__global__ void __launch_bounds__(128)
k_lstm_rec(const float* __restrict__ pre,
           const float* __restrict__ w_hh,
           _Float16* __restrict__ hout) {
    __shared__ _Float16 hls[16][72];     // h_t tile, padded rows

    const int tid  = threadIdx.x;
    const int wv   = tid >> 5;
    const int lane = tid & 31;
    const int g    = lane >> 4;
    const int n    = lane & 15;
    const int bt   = blockIdx.x;
    const int m0   = bt * 16;            // batch-tile base row

    // W_hh B-fragments (f32 global -> f16 regs), fixed for all steps.
    V16 bw[4][2];
    #pragma unroll
    for (int q = 0; q < 4; ++q) {
        const int col = q * 64 + wv * 16 + n;
        #pragma unroll
        for (int kf = 0; kf < 2; ++kf) {
            const float* wr = w_hh + col * HH + kf * 32 + g * 16;
            #pragma unroll
            for (int h = 0; h < 16; ++h) bw[q][kf].e[h] = (_Float16)wr[h];
        }
    }

    // Cell state in registers: element (M = r + 8g, N = wv*16 + n).
    float c8[8];
    #pragma unroll
    for (int r = 0; r < 8; ++r) c8[r] = 0.0f;

    // h_0 = 0
    for (int i = tid; i < 16 * 72; i += 128) ((_Float16*)hls)[i] = (_Float16)0.0f;

    // Prefetch pre tiles for t=0 and t=1 into ping-pong register sets.
    v8f p0[4], p1[4];
    #pragma unroll
    for (int q = 0; q < 4; ++q) {
        p0[q] = ld_pre(pre, bt, wv, lane, 0, q);
        p1[q] = ld_pre(pre, bt, wv, lane, 1, q);
    }

    auto step = [&](int t, v8f (&pc)[4]) {
        __syncthreads();   // (a) previous step's h writes visible

        // A fragments = h_{t-1} from LDS (row m = n).
        V16 a[2];
        #pragma unroll
        for (int kf = 0; kf < 2; ++kf) {
            a[kf].h2[0] = *(const v8h*)&hls[n][kf * 32 + g * 16];
            a[kf].h2[1] = *(const v8h*)&hls[n][kf * 32 + g * 16 + 8];
        }

        // gates = pre_t + h_{t-1} @ W_hh^T; pre_t enters as the WMMA C
        // operand directly. Reload pc with step t+2 immediately after use.
        v8f acc[4];
        #pragma unroll
        for (int q = 0; q < 4; ++q) {
            acc[q] = __builtin_amdgcn_wmma_f32_16x16x32_f16(
                        false, a[0].v, false, bw[q][0].v, (short)0, pc[q],
                        false, false);
            if (t + 2 < TT) pc[q] = ld_pre(pre, bt, wv, lane, t + 2, q);
            acc[q] = __builtin_amdgcn_wmma_f32_16x16x32_f16(
                        false, a[1].v, false, bw[q][1].v, (short)0, acc[q],
                        false, false);
        }

        // Elementwise LSTM cell (gate order i, f, g, o).
        _Float16 hnew[8];
        #pragma unroll
        for (int r = 0; r < 8; ++r) {
            float iv = fsigmoid(acc[0][r]);
            float fv = fsigmoid(acc[1][r]);
            float gv = ftanh(acc[2][r]);
            float ov = fsigmoid(acc[3][r]);
            c8[r] = fv * c8[r] + iv * gv;
            hnew[r] = (_Float16)(ov * ftanh(c8[r]));
        }

        __syncthreads();   // (b) all A reads of h_{t-1} done -> h writable

        #pragma unroll
        for (int r = 0; r < 8; ++r) {
            const int m = r + 8 * g;
            hls[m][wv * 16 + n] = hnew[r];
            hout[((size_t)t * BB + m0 + m) * HH + wv * 16 + n] = hnew[r];
        }
    };

    #pragma unroll 1
    for (int t = 0; t < TT; t += 2) {
        step(t, p0);
        step(t + 1, p1);
    }
}

// ---------------------------------------------------------------------------
// Kernel 4: out[b, o] = softplus(h2[T-1, b, :] . fc_w[o, :] + fc_b[o])
// ---------------------------------------------------------------------------
__global__ void __launch_bounds__(256)
k_fc_softplus(const _Float16* __restrict__ h2,
              const float* __restrict__ fw,
              const float* __restrict__ fb,
              float* __restrict__ out) {
    __shared__ float wls[OO * HH];
    __shared__ float bls[OO];
    const int tid = threadIdx.x;
    for (int i = tid; i < OO * HH; i += 256) wls[i] = fw[i];
    if (tid < OO) bls[tid] = fb[tid];
    __syncthreads();

    const int idx = blockIdx.x * 256 + tid;   // b*32 + o
    const int b = idx >> 5;
    const int o = idx & 31;
    const _Float16* hr = h2 + ((size_t)(TT - 1) * BB + b) * HH;
    float s = bls[o];
    #pragma unroll
    for (int k = 0; k < HH; ++k) s += (float)hr[k] * wls[o * HH + k];
    out[idx] = (s > 20.0f) ? s : log1pf(__expf(s));
}

// ---------------------------------------------------------------------------
// Launch sequence
// ---------------------------------------------------------------------------
extern "C" void kernel_launch(void* const* d_in, const int* in_sizes, int n_in,
                              void* d_out, int out_size, void* d_ws, size_t ws_size,
                              hipStream_t stream) {
    const float* x     = (const float*)d_in[0];
    const float* w_ih0 = (const float*)d_in[1];
    const float* w_hh0 = (const float*)d_in[2];
    const float* b_ih0 = (const float*)d_in[3];
    const float* b_hh0 = (const float*)d_in[4];
    const float* w_ih1 = (const float*)d_in[5];
    const float* w_hh1 = (const float*)d_in[6];
    const float* b_ih1 = (const float*)d_in[7];
    const float* b_hh1 = (const float*)d_in[8];
    const float* fc_w  = (const float*)d_in[9];
    const float* fc_b  = (const float*)d_in[10];

    char* ws = (char*)d_ws;
    const size_t ACT_BYTES = (size_t)BB * TT * DD * sizeof(_Float16); // 64 MB
    _Float16* xf  = (_Float16*)(ws);                    // (T,B,64) f16
    _Float16* h1  = (_Float16*)(ws + ACT_BYTES);        // (T,B,64) f16
    _Float16* h2  = (_Float16*)(ws + 2 * ACT_BYTES);    // (T,B,64) f16
    float*    pre = (float*)(ws + 3 * ACT_BYTES);       // fragment f32, 512 MB

    const int nact = BB * TT * DD;                      // 33.5M elements
    k_cvt_transpose<<<nact / 256, 256, 0, stream>>>(x, xf);

    const int gemm_blocks = (BB * TT) / 128;            // 4096

    // Layer 0
    k_gemm_pre<<<gemm_blocks, 256, 0, stream>>>(xf, w_ih0, b_ih0, b_hh0, pre);
    k_lstm_rec<<<BB / 16, 128, 0, stream>>>(pre, w_hh0, h1);

    // Layer 1
    k_gemm_pre<<<gemm_blocks, 256, 0, stream>>>(h1, w_ih1, b_ih1, b_hh1, pre);
    k_lstm_rec<<<BB / 16, 128, 0, stream>>>(pre, w_hh1, h2);

    // Head
    k_fc_softplus<<<(BB * OO) / 256, 256, 0, stream>>>(h2, fc_w, fc_b,
                                                       (float*)d_out);
}
